// MADT_14164802142922
// MI455X (gfx1250) — matrix operand order repaired and checked by reading, not compiled, not run on hardware
//
#include <hip/hip_runtime.h>
#include <math.h>

// ---------------------------------------------------------------------------
// MADT transformer forward for MI455X (gfx1250, wave32, WMMA).
// All GEMM-shaped math runs through v_wmma_f32_16x16x32_f16 (f16 in, f32 acc).
// Fragment operands are staged in LDS so every lane's 16 halves are contiguous
// -> ds_load_b128 pairs. Transposed tiles are transposed in registers so all
// LDS stores are packed b64, and staging loops are exact-trip (no exec masks).
// ---------------------------------------------------------------------------

typedef _Float16 f16;
typedef __attribute__((ext_vector_type(4)))  _Float16 v4h;
typedef __attribute__((ext_vector_type(8)))  _Float16 v8h;
typedef __attribute__((ext_vector_type(16))) _Float16 v16h;
typedef __attribute__((ext_vector_type(8)))  float    v8f;

__device__ __forceinline__ v8f wmma16x16x32(v16h a, v16h b, v8f c) {
  // (neg_a, A, neg_b, B, c_mod, C, reuse_a, reuse_b)
  return __builtin_amdgcn_wmma_f32_16x16x32_f16(false, a, false, b, (short)0, c,
                                                false, false);
}

// A-fragment (16x32 f16) from a row-major [M][K] LDS tile.
// Lane (hh,m): elements e=0..7 -> K = k0+hh*8+e ; e=8..15 -> K = k0+16+hh*8+(e-8).
// Both chunks are 16B-aligned when ld%8==0 and k0%8==0  -> two ds_load_b128.
__device__ __forceinline__ v16h load_a_frag(const f16* lds, int row0, int k0,
                                            int ld) {
  const int lane = threadIdx.x & 31;
  const int hh = lane >> 4;
  const int m = lane & 15;
  const f16* base = lds + (row0 + m) * ld + k0 + hh * 8;
  const v8h lo = *(const v8h*)(base);
  const v8h hi = *(const v8h*)(base + 16);
  return __builtin_shufflevector(lo, hi, 0, 1, 2, 3, 4, 5, 6, 7, 8, 9, 10, 11,
                                 12, 13, 14, 15);
}

// B-fragment (32x16 f16) from a TRANSPOSED [N][K] LDS tile.
// Lane (kh,n): e=0..7 -> K = k0+kh*8+e ; e=8..15 -> K = k0+16+kh*8+(e-8)
// -> two ds_load_b128 (ld%8==0, k0%8==0).
__device__ __forceinline__ v16h load_bt_frag(const f16* lds, int k0, int col0,
                                             int ld) {
  const int lane = threadIdx.x & 31;
  const int kh = lane >> 4;
  const int n = lane & 15;
  const f16* base = lds + (col0 + n) * ld + k0 + kh * 8;
  const v8h lo = *(const v8h*)(base);
  const v8h hi = *(const v8h*)(base + 16);
  return __builtin_shufflevector(lo, hi, 0, 1, 2, 3, 4, 5, 6, 7, 8, 9, 10, 11,
                                 12, 13, 14, 15);
}

__device__ __forceinline__ v4h cvt4(float x, float y, float z, float w) {
  v4h r;
  r[0] = (f16)x; r[1] = (f16)y; r[2] = (f16)z; r[3] = (f16)w;
  return r;
}

__device__ __forceinline__ float gelu_exact(float x) {
  return 0.5f * x * (1.0f + erff(x * 0.70710678118654752f));
}

// ---------------------------------------------------------------------------
// Tiled WMMA GEMM: Y[M,N] = epilogue( X[M,K] @ W[K,N] + bias )
// resMode: 0 none, 1 full residual res[M,N], 2 positional res[(m%Tpos),N]
// act:     0 none, 1 gelu(exact), 2 tanh        (activation BEFORE residual)
// M % 64 == 0, N % 64 == 0, K % 32 == 0 (true for every call here).
// ---------------------------------------------------------------------------
#define GBM 64
#define GBN 64
#define GBK 32
#define XS_LD 40  // halves; %8==0 -> 16B rows
#define WT_LD 40

__global__ __launch_bounds__(256) void madt_gemm(
    const float* __restrict__ X, const float* __restrict__ W,
    const float* __restrict__ bias, const float* __restrict__ res, int resMode,
    int act, float* __restrict__ Y, int M, int N, int K, int Tpos) {
  __shared__ __align__(16) f16 Xs[GBM * XS_LD];        // [m][k]
  __shared__ __align__(16) f16 Wt[GBN * WT_LD];        // transposed: [n][k]
  const int tid = threadIdx.x;
  const int bm = blockIdx.y * GBM;
  const int bn = blockIdx.x * GBN;
  const int w = tid >> 5;
  const int wrow = (w & 3) * 16;
  const int wcol = (w >> 2) * 32;
  v8f acc0 = {};
  v8f acc1 = {};

  for (int k0 = 0; k0 < K; k0 += GBK) {
    // stage X tile (64x32): float4 global reads -> packed v4h LDS stores
#pragma unroll
    for (int it = 0; it < 2; ++it) {
      const int i = tid + it * 256;
      const int r = i >> 3;
      const int c = (i & 7) * 4;
      const float4 f =
          *(const float4*)(X + (size_t)(bm + r) * K + (k0 + c));
      *(v4h*)(&Xs[r * XS_LD + c]) = cvt4(f.x, f.y, f.z, f.w);
    }
    // stage W tile (32x64) transposed IN REGISTERS: each thread owns column n
    // and a K-quad; 4 coalesced b32 reads -> one packed ds_store_b64.
#pragma unroll
    for (int it = 0; it < 2; ++it) {
      const int i = tid + it * 256;
      const int n = i & 63;
      const int kq = (i >> 6) * 4;  // 0,4,...,28
      const float* wp = W + (size_t)(k0 + kq) * N + (bn + n);
      *(v4h*)(&Wt[n * WT_LD + kq]) =
          cvt4(wp[0], wp[(size_t)N], wp[(size_t)2 * N], wp[(size_t)3 * N]);
    }
    if (k0 + GBK < K) {  // hint next tiles into cache (global_prefetch_b8)
      __builtin_prefetch(X + (size_t)(bm + (tid & 63)) * K + (k0 + GBK), 0, 1);
      __builtin_prefetch(W + (size_t)(k0 + GBK + (tid & 31)) * N + bn, 0, 1);
    }
    __syncthreads();

    const v16h af = load_a_frag(Xs, wrow, 0, XS_LD);
    const v16h b0 = load_bt_frag(Wt, 0, wcol, WT_LD);
    const v16h b1 = load_bt_frag(Wt, 0, wcol + 16, WT_LD);
    acc0 = wmma16x16x32(af, b0, acc0);
    acc1 = wmma16x16x32(af, b1, acc1);
    __syncthreads();
  }

  // epilogue straight out of the C-fragment layout
  const int lane = tid & 31;
  const int rl = (lane >> 4) << 3;  // +8 rows for upper lane half
  const int cl = lane & 15;
#pragma unroll
  for (int i = 0; i < 8; ++i) {
    const int gr = bm + wrow + i + rl;
#pragma unroll
    for (int fr = 0; fr < 2; ++fr) {
      const int gc = bn + wcol + fr * 16 + cl;
      float v = (fr == 0) ? acc0[i] : acc1[i];
      if (bias) v += bias[gc];
      if (act == 1)
        v = gelu_exact(v);
      else if (act == 2)
        v = tanhf(v);
      if (resMode == 1)
        v += res[(size_t)gr * N + gc];
      else if (resMode == 2)
        v += res[(size_t)(gr % Tpos) * N + gc];
      Y[(size_t)gr * N + gc] = v;
    }
  }
}

// ---------------------------------------------------------------------------
// Fused flash-style attention for one (b, h, 32-query-row block).
// Q,K,V are [B,T,C] fp32 with head h in columns h*64..h*64+63.
// Scale 1/sqrt(64) folded into the Q load. WMMA for QK^T and PV.
// B-operand staging:  S = Q @ K^T needs (K^T)^T = K natural [key][d];
//                     PV = P @ V  needs V^T            = Vt [d][key].
// ---------------------------------------------------------------------------
#define AQ 32
#define AK 64
#define DH 64
#define QLD 72   // %8==0
#define KLD 72
#define VLD 72
#define PLD 72
#define SLD 68
#define OLD 68

__global__ __launch_bounds__(256) void madt_attn(
    const float* __restrict__ Q, const float* __restrict__ K,
    const float* __restrict__ V, float* __restrict__ Y, int Tq, int Tk,
    int causal) {
  const int Cc = 512;
  __shared__ __align__(16) f16 Qs[AQ * QLD];   // [q][d]
  __shared__ __align__(16) f16 Ks[AK * KLD];   // [key][d] (transpose of K^T)
  __shared__ __align__(16) f16 Vt[DH * VLD];   // [d][key] (transpose of V)
  __shared__ __align__(16) f16 Pf[AQ * PLD];   // [q][key]
  __shared__ float Sf[AQ * SLD];
  __shared__ float Of[AQ * OLD];
  __shared__ float rowm[AQ], rowl[AQ];
  __shared__ float redm[AQ * 8], reds[AQ * 8];

  const int tid = threadIdx.x;
  const int b = blockIdx.z;
  const int h = blockIdx.y;
  const int qb = blockIdx.x;
  const int w = tid >> 5;
  const int lane = tid & 31;
  const int wrow = (w & 1) * 16;
  const int wcol = (w >> 1) * 16;
  const int rl = (lane >> 4) << 3;
  const int cl = lane & 15;

  const size_t qbase = ((size_t)(b * Tq + qb * AQ)) * Cc + h * DH;

  // load + pre-scale Q tile (float4 reads, v4h stores)
#pragma unroll
  for (int it = 0; it < 2; ++it) {
    const int i = tid + it * 256;
    const int r = i >> 4;
    const int d = (i & 15) * 4;
    const float4 f = *(const float4*)(Q + qbase + (size_t)r * Cc + d);
    *(v4h*)(&Qs[r * QLD + d]) =
        cvt4(f.x * 0.125f, f.y * 0.125f, f.z * 0.125f, f.w * 0.125f);
  }
  // init O (exact 32x64 region) and row stats
#pragma unroll
  for (int it = 0; it < 8; ++it) {
    const int i = tid + it * 256;
    Of[(i >> 6) * OLD + (i & 63)] = 0.0f;
  }
  if (tid < AQ) {
    rowm[tid] = -3.0e38f;
    rowl[tid] = 0.0f;
  }
  __syncthreads();

  const int nkb = causal ? ((qb * AQ + AQ - 1) / AK + 1) : (Tk / AK);
  for (int kb = 0; kb < nkb; ++kb) {
    const size_t kbase = ((size_t)(b * Tk + kb * AK)) * Cc + h * DH;
    // stage K tile (natural [key][d]): float4 reads -> packed v4h stores
#pragma unroll
    for (int it = 0; it < 4; ++it) {
      const int i = tid + it * 256;
      const int r = i >> 4;          // key within block
      const int d = (i & 15) * 4;
      const float4 fk = *(const float4*)(K + kbase + (size_t)r * Cc + d);
      *(v4h*)(&Ks[r * KLD + d]) = cvt4(fk.x, fk.y, fk.z, fk.w);
    }
    // stage V tile transposed IN REGISTERS: thread owns d and a key-quad;
    // 4 coalesced b32 reads -> one packed ds_store_b64.
#pragma unroll
    for (int it = 0; it < 4; ++it) {
      const int i = tid + it * 256;
      const int d = i & 63;
      const int kq = (i >> 6) * 4;  // 0,4,...,60
      const float* vp = V + kbase + (size_t)kq * Cc + d;
      *(v4h*)(&Vt[d * VLD + kq]) =
          cvt4(vp[0], vp[(size_t)Cc], vp[(size_t)2 * Cc], vp[(size_t)3 * Cc]);
    }
    __syncthreads();

    // S = Q @ K^T   (each wave: one 16x16 tile of the 32x64 block)
    {
      v8f sacc = {};
#pragma unroll
      for (int ks = 0; ks < DH; ks += 32) {
        const v16h a = load_a_frag(Qs, wrow, ks, QLD);
        const v16h bb = load_bt_frag(Ks, ks, wcol, KLD);
        sacc = wmma16x16x32(a, bb, sacc);
      }
#pragma unroll
      for (int i = 0; i < 8; ++i) {
        const int r = wrow + i + rl;
        const int c = wcol + cl;
        float sv = sacc[i];
        if (causal && (kb * AK + c) > (qb * AQ + r)) sv = -3.0e38f;
        Sf[r * SLD + c] = sv;
      }
    }
    __syncthreads();

    // online softmax update: 256 threads = 32 rows x 8 segments of 8 cols
    {
      const int row = tid >> 3;
      const int seg = tid & 7;
      float lm = -3.0e38f;
#pragma unroll
      for (int j = 0; j < 8; ++j)
        lm = fmaxf(lm, Sf[row * SLD + seg * 8 + j]);
      redm[row * 8 + seg] = lm;
      __syncthreads();
      const float oldm = rowm[row];
      float newm = oldm;
#pragma unroll
      for (int s = 0; s < 8; ++s) newm = fmaxf(newm, redm[row * 8 + s]);
      const float alpha = __expf(oldm - newm);
      float lsum = 0.0f;
#pragma unroll
      for (int j = 0; j < 8; ++j) {
        const float p = __expf(Sf[row * SLD + seg * 8 + j] - newm);
        Pf[row * PLD + seg * 8 + j] = (f16)p;
        lsum += p;
      }
      reds[row * 8 + seg] = lsum;
#pragma unroll
      for (int j = 0; j < 8; ++j) Of[row * OLD + seg * 8 + j] *= alpha;
      __syncthreads();
      if (seg == 0) {
        float t = 0.0f;
#pragma unroll
        for (int s = 0; s < 8; ++s) t += reds[row * 8 + s];
        rowl[row] = rowl[row] * alpha + t;
        rowm[row] = newm;
      }
    }
    __syncthreads();

    // PV = P @ V  (reuse Sf as the fp32 staging tile)
    {
      v8f pacc = {};
#pragma unroll
      for (int ks = 0; ks < AK; ks += 32) {
        const v16h a = load_a_frag(Pf, wrow, ks, PLD);
        const v16h bb = load_bt_frag(Vt, ks, wcol, VLD);
        pacc = wmma16x16x32(a, bb, pacc);
      }
#pragma unroll
      for (int i = 0; i < 8; ++i) {
        const int r = wrow + i + rl;
        const int c = wcol + cl;
        Sf[r * SLD + c] = pacc[i];
      }
    }
    __syncthreads();
    {
      const int row = tid >> 3;
      const int seg = tid & 7;
#pragma unroll
      for (int j = 0; j < 8; ++j)
        Of[row * OLD + seg * 8 + j] += Sf[row * SLD + seg * 8 + j];
    }
    __syncthreads();
  }

  // normalize and write out (float4 global stores)
#pragma unroll
  for (int it = 0; it < 2; ++it) {
    const int i = tid + it * 256;
    const int r = i >> 4;
    const int d = (i & 15) * 4;
    const float inv = 1.0f / rowl[r];
    float4 o;
    o.x = Of[r * OLD + d + 0] * inv;
    o.y = Of[r * OLD + d + 1] * inv;
    o.z = Of[r * OLD + d + 2] * inv;
    o.w = Of[r * OLD + d + 3] * inv;
    *(float4*)(Y + qbase + (size_t)r * Cc + d) = o;
  }
}

// ---------------------------------------------------------------------------
// LayerNorm over C=512 (one block per row, 256 threads, 2 elems/thread).
// ---------------------------------------------------------------------------
__global__ __launch_bounds__(256) void madt_ln(const float* __restrict__ X,
                                               const float* __restrict__ g,
                                               const float* __restrict__ be,
                                               float* __restrict__ Y) {
  const int C = 512;
  __shared__ float red[8];
  __shared__ float stat[2];
  const int row = blockIdx.x;
  const int tid = threadIdx.x;
  const float* xr = X + (size_t)row * C;
  const float v0 = xr[tid];
  const float v1 = xr[tid + 256];
  float s = v0 + v1;
#pragma unroll
  for (int o = 16; o > 0; o >>= 1) s += __shfl_xor(s, o, 32);
  if ((tid & 31) == 0) red[tid >> 5] = s;
  __syncthreads();
  if (tid == 0) {
    float t = 0.0f;
    for (int i = 0; i < 8; ++i) t += red[i];
    stat[0] = t / C;
  }
  __syncthreads();
  const float mean = stat[0];
  const float d0 = v0 - mean, d1 = v1 - mean;
  float q = d0 * d0 + d1 * d1;
#pragma unroll
  for (int o = 16; o > 0; o >>= 1) q += __shfl_xor(q, o, 32);
  __syncthreads();
  if ((tid & 31) == 0) red[tid >> 5] = q;
  __syncthreads();
  if (tid == 0) {
    float t = 0.0f;
    for (int i = 0; i < 8; ++i) t += red[i];
    stat[1] = rsqrtf(t / C + 1e-5f);
  }
  __syncthreads();
  const float inv = stat[1];
  Y[(size_t)row * C + tid] = d0 * inv * g[tid] + be[tid];
  Y[(size_t)row * C + tid + 256] = d1 * inv * g[tid + 256] + be[tid + 256];
}

// a[bt, c] = tanh(act_tab[actions[bt], c]) + pos[bt % T, c]
__global__ __launch_bounds__(256) void madt_embed_act(
    const int* __restrict__ actions, const float* __restrict__ act_tab,
    const float* __restrict__ pos, float* __restrict__ A, int T) {
  const int C = 512;
  const int idx = blockIdx.x * 256 + threadIdx.x;
  const int bt = idx >> 9;
  const int c = idx & 511;
  const int t = bt & (T - 1);
  A[idx] = tanhf(act_tab[(size_t)actions[bt] * C + c]) + pos[(size_t)t * C + c];
}

// ---------------------------------------------------------------------------
// Host orchestration
// ---------------------------------------------------------------------------
extern "C" void kernel_launch(void* const* d_in, const int* in_sizes, int n_in,
                              void* d_out, int out_size, void* d_ws,
                              size_t ws_size, hipStream_t stream) {
  (void)in_sizes; (void)n_in; (void)out_size; (void)ws_size;
  constexpr int B = 4, T = 1024, C = 512, H = 8, L = 4, V = 64, S = 256;
  constexpr int M = B * T;       // 4096 token rows
  constexpr int C2 = 2 * C;      // 1024

  const float* states    = (const float*)d_in[0];
  const int*   actions   = (const int*)d_in[1];
  const float* ws_enc    = (const float*)d_in[2];
  const float* bs_enc    = (const float*)d_in[3];
  const float* act_tab   = (const float*)d_in[4];
  const float* pos       = (const float*)d_in[5];
  const float* enc_ln_s  = (const float*)d_in[6];
  const float* enc_ln_b  = (const float*)d_in[7];
  const float* enc_attn_w = (const float*)d_in[8];
  const float* enc_attn_b = (const float*)d_in[9];
  const float* enc_mlp_w1 = (const float*)d_in[10];
  const float* enc_mlp_b1 = (const float*)d_in[11];
  const float* enc_mlp_w2 = (const float*)d_in[12];
  const float* enc_mlp_b2 = (const float*)d_in[13];
  const float* dec_ln_s  = (const float*)d_in[14];
  const float* dec_ln_b  = (const float*)d_in[15];
  const float* dec_self_w = (const float*)d_in[16];
  const float* dec_self_b = (const float*)d_in[17];
  const float* dec_cross_w = (const float*)d_in[18];
  const float* dec_cross_b = (const float*)d_in[19];
  const float* dec_mlp_w1 = (const float*)d_in[20];
  const float* dec_mlp_b1 = (const float*)d_in[21];
  const float* dec_mlp_w2 = (const float*)d_in[22];
  const float* dec_mlp_b2 = (const float*)d_in[23];
  const float* lnf_s     = (const float*)d_in[24];
  const float* lnf_b     = (const float*)d_in[25];
  const float* head_w    = (const float*)d_in[26];
  float* out = (float*)d_out;

  // workspace carve (~72 MB fp32)
  float* ws = (float*)d_ws;
  size_t off = 0;
  auto carve = [&](size_t n) { float* p = ws + off; off += n; return p; };
  float* x   = carve((size_t)M * C);   // encoder stream / memory
  float* a   = carve((size_t)M * C);   // decoder stream
  float* h   = carve((size_t)M * C);   // LN output
  float* qb  = carve((size_t)M * C);
  float* kb  = carve((size_t)M * C);
  float* vb  = carve((size_t)M * C);
  float* yb  = carve((size_t)M * C);
  float* hid = carve((size_t)M * C2);

  auto gemm = [&](const float* X, const float* W, const float* bias,
                  const float* res, int resMode, int act, float* Y, int Mm,
                  int Nn, int Kk) {
    dim3 g(Nn / 64, Mm / 64);
    madt_gemm<<<g, dim3(256), 0, stream>>>(X, W, bias, res, resMode, act, Y,
                                           Mm, Nn, Kk, T);
  };
  auto ln = [&](const float* X, const float* gs, const float* gb, float* Y) {
    madt_ln<<<dim3(M), dim3(256), 0, stream>>>(X, gs, gb, Y);
  };
  auto attn = [&](const float* Qp, const float* Kp, const float* Vp, float* Yp,
                  int causal) {
    dim3 g(T / AQ, H, B);
    madt_attn<<<g, dim3(256), 0, stream>>>(Qp, Kp, Vp, Yp, T, T, causal);
  };

  // ---- Encoder ----
  // x = tanh(states @ ws_enc + bs_enc) + pos
  gemm(states, ws_enc, bs_enc, pos, /*res=pos*/ 2, /*tanh*/ 2, x, M, C, S);
  for (int i = 0; i < L; ++i) {
    const float* w4 = enc_attn_w + (size_t)(i * 4) * C * C;
    const float* b4 = enc_attn_b + (size_t)(i * 4) * C;
    ln(x, enc_ln_s + (size_t)(i * 2 + 0) * C, enc_ln_b + (size_t)(i * 2 + 0) * C, h);
    gemm(h, w4 + 0 * (size_t)C * C, b4 + 0 * C, nullptr, 0, 0, qb, M, C, C);
    gemm(h, w4 + 1 * (size_t)C * C, b4 + 1 * C, nullptr, 0, 0, kb, M, C, C);
    gemm(h, w4 + 2 * (size_t)C * C, b4 + 2 * C, nullptr, 0, 0, vb, M, C, C);
    attn(qb, kb, vb, yb, /*causal*/ 1);
    gemm(yb, w4 + 3 * (size_t)C * C, b4 + 3 * C, x, 1, 0, x, M, C, C);
    ln(x, enc_ln_s + (size_t)(i * 2 + 1) * C, enc_ln_b + (size_t)(i * 2 + 1) * C, h);
    gemm(h, enc_mlp_w1 + (size_t)i * C * C2, enc_mlp_b1 + (size_t)i * C2,
         nullptr, 0, /*gelu*/ 1, hid, M, C2, C);
    gemm(hid, enc_mlp_w2 + (size_t)i * C2 * C, enc_mlp_b2 + (size_t)i * C, x, 1,
         0, x, M, C, C2);
  }
  // mem == x (kept intact through decoding)

  // ---- Decoder ----
  madt_embed_act<<<dim3((M * C) / 256), dim3(256), 0, stream>>>(actions,
                                                                act_tab, pos,
                                                                a, T);
  for (int i = 0; i < L; ++i) {
    const float* ws4 = dec_self_w + (size_t)(i * 4) * C * C;
    const float* bs4 = dec_self_b + (size_t)(i * 4) * C;
    const float* wc4 = dec_cross_w + (size_t)(i * 4) * C * C;
    const float* bc4 = dec_cross_b + (size_t)(i * 4) * C;
    // causal self-attention
    ln(a, dec_ln_s + (size_t)(i * 3 + 0) * C, dec_ln_b + (size_t)(i * 3 + 0) * C, h);
    gemm(h, ws4 + 0 * (size_t)C * C, bs4 + 0 * C, nullptr, 0, 0, qb, M, C, C);
    gemm(h, ws4 + 1 * (size_t)C * C, bs4 + 1 * C, nullptr, 0, 0, kb, M, C, C);
    gemm(h, ws4 + 2 * (size_t)C * C, bs4 + 2 * C, nullptr, 0, 0, vb, M, C, C);
    attn(qb, kb, vb, yb, 1);
    gemm(yb, ws4 + 3 * (size_t)C * C, bs4 + 3 * C, a, 1, 0, a, M, C, C);
    // cross-attention to memory (K/V from raw mem)
    ln(a, dec_ln_s + (size_t)(i * 3 + 1) * C, dec_ln_b + (size_t)(i * 3 + 1) * C, h);
    gemm(h, wc4 + 0 * (size_t)C * C, bc4 + 0 * C, nullptr, 0, 0, qb, M, C, C);
    gemm(x, wc4 + 1 * (size_t)C * C, bc4 + 1 * C, nullptr, 0, 0, kb, M, C, C);
    gemm(x, wc4 + 2 * (size_t)C * C, bc4 + 2 * C, nullptr, 0, 0, vb, M, C, C);
    attn(qb, kb, vb, yb, 0);
    gemm(yb, wc4 + 3 * (size_t)C * C, bc4 + 3 * C, a, 1, 0, a, M, C, C);
    // MLP
    ln(a, dec_ln_s + (size_t)(i * 3 + 2) * C, dec_ln_b + (size_t)(i * 3 + 2) * C, h);
    gemm(h, dec_mlp_w1 + (size_t)i * C * C2, dec_mlp_b1 + (size_t)i * C2,
         nullptr, 0, 1, hid, M, C2, C);
    gemm(hid, dec_mlp_w2 + (size_t)i * C2 * C, dec_mlp_b2 + (size_t)i * C, a, 1,
         0, a, M, C, C2);
  }

  // ---- Head ----
  ln(a, lnf_s, lnf_b, h);
  gemm(h, head_w, nullptr, nullptr, 0, 0, out, M, V, C);
}
